// MemTransformerLM_50414326120576
// MI455X (gfx1250) — matrix operand order repaired and checked by reading, not compiled, not run on hardware
//
#include <hip/hip_runtime.h>
#include <hip/hip_bf16.h>

// ---------------- model constants ----------------
#define L_      12
#define NH_     12
#define DH_     64
#define D_      768
#define DI_     3072
#define V_      10000
#define QLEN_   512
#define MLEN_   512
#define CLEN_   32
#define BSZ_    4
#define KLEN_   (CLEN_ + MLEN_ + QLEN_)   // 1056
#define HDIM_   (NH_ * DH_)               // 768
#define QKV3_   (3 * HDIM_)               // 2304
#define ROWS_Q  (QLEN_ * BSZ_)            // 2048
#define ROWS_K  (KLEN_ * BSZ_)            // 4224

typedef unsigned short u16b;              // bf16 bit pattern
typedef __attribute__((ext_vector_type(16))) __bf16    v16bf;
typedef __attribute__((ext_vector_type(8)))  float     v8f;
typedef __attribute__((ext_vector_type(8)))  unsigned  v8u;
typedef __attribute__((ext_vector_type(4)))  int       v4i;

#if defined(__gfx1250__) && __has_builtin(__builtin_amdgcn_global_load_async_to_lds_b128)
#define HAVE_ASYNC_LDS 1
// builtin signature: (v4i addrspace(1)*, v4i addrspace(3)*, imm offset, imm cpol)
#define GPTR(p) ((__attribute__((address_space(1))) v4i*)(void*)(p))
#define LPTR(p) ((__attribute__((address_space(3))) v4i*)(p))
#else
#define HAVE_ASYNC_LDS 0
#endif

// f32 -> bf16 bits (RTNE)
__device__ __forceinline__ u16b f2bfu(float f) {
  unsigned u = __float_as_uint(f);
  u += 0x7FFFu + ((u >> 16) & 1u);
  return (u16b)(u >> 16);
}

__device__ __forceinline__ v8f wmma_bf16(v16bf a, v16bf b, v8f c) {
  return __builtin_amdgcn_wmma_f32_16x16x32_bf16(false, a, false, b, (short)0, c, false, false);
}

// ---------------- generic batched bf16-WMMA GEMM (all-NN, bf16 operands) ----
// C[z][M,N] = A[z][M,K] * B[z][K,N]  (+bias, optional ReLU); A,B bf16; C f32 or bf16.
// batch offset = (z/zdiv)*s?0 + (z%zdiv)*s?1
// Requirements (hold for every call site): K % 32 == 0; N % 8 == 0; lda/ldb and all
// base/stride offsets are multiples of 8 elements (16B) -> unconditional B128 copies.
// OOB rows/cols are index-clamped: garbage only reaches C rows/cols that are never stored.
// Block: 256 threads = 8 waves; tile 128(M) x 64(N), wave sub-tile 32x32, K-step 32.
__global__ __launch_bounds__(256) void gemm_wmma(
    const u16b* __restrict__ A, long long sA0, long long sA1, int lda,
    const u16b* __restrict__ B, long long sB0, long long sB1, int ldb,
    void* __restrict__ Cv, long long sC0, long long sC1, int ldc,
    int M, int N, int K, int zdiv,
    const float* __restrict__ bias, int relu, int cbf)
{
  // bf16 tiles as dwords; row strides 20/36 dwords keep every 16B chunk 16B-aligned
  __shared__ unsigned As[128][20];  // 128 x 32 bf16 (A tile)
  __shared__ unsigned Bs[32][36];   //  32 x 64 bf16 (B tile)

  const int z = blockIdx.z, zo = z / zdiv, zi = z % zdiv;
  A += zo * sA0 + zi * sA1;
  B += zo * sB0 + zi * sB1;

  const int tid  = threadIdx.x;
  const int lane = tid & 31;
  const int wave = tid >> 5;              // 8 waves: 4x2 grid of 32x32 sub-tiles
  const int m0   = blockIdx.y * 128;
  const int n0   = blockIdx.x * 64;
  const int wm   = (wave >> 1) << 5;      // 0/32/64/96
  const int wn   = (wave & 1) << 5;       // 0/32
  const int half = lane >> 4;
  const int mrow = lane & 15;

  // one 16B chunk = 8 bf16
  auto cpA = [&](int k0) {                // A tile: 512 chunks, 2 per thread
    #pragma unroll
    for (int i = 0; i < 2; ++i) {
      int idx = tid + (i << 8);
      int r = idx >> 2, ch = idx & 3;
      int gm = m0 + r;
      int gms = (gm < M) ? gm : 0;        // clamp (stays in-bounds)
      const u16b* src = A + (long long)gms * lda + k0 + (ch << 3);
      unsigned* dst = &As[r][ch << 2];
#if HAVE_ASYNC_LDS
      __builtin_amdgcn_global_load_async_to_lds_b128(GPTR(src), LPTR(dst), 0, 0);
#else
      *reinterpret_cast<uint4*>(dst) = *reinterpret_cast<const uint4*>(src);
#endif
    }
  };
  auto cpB = [&](int k0) {                // B tile: 256 chunks, 1 per thread
    int r = tid >> 3, ch = tid & 7;
    int gn = n0 + (ch << 3);
    int gns = (gn < N) ? gn : 0;          // N % 8 == 0 -> whole chunk in/out together
    const u16b* src = B + (long long)(k0 + r) * ldb + gns;
    unsigned* dst = &Bs[r][ch << 2];
#if HAVE_ASYNC_LDS
    __builtin_amdgcn_global_load_async_to_lds_b128(GPTR(src), LPTR(dst), 0, 0);
#else
    *reinterpret_cast<uint4*>(dst) = *reinterpret_cast<const uint4*>(src);
#endif
  };

  v8f acc00 = {}, acc01 = {}, acc10 = {}, acc11 = {};

  for (int k0 = 0; k0 < K; k0 += 32) {
    cpA(k0); cpB(k0);
#if HAVE_ASYNC_LDS
    asm volatile("s_wait_asynccnt 0x0" ::: "memory");
#endif
    __syncthreads();

    // CDNA5 wave32 operand gathers (dword-pair granularity)
    v8u au0, au1, bu0, bu1;
    #pragma unroll
    for (int v = 0; v < 8; ++v) {
      // A 16x32 layout: lanes 0-15 -> K 0..7,16..23 ; lanes 16-31 -> K 8..15,24..31
      int ui = v + 4 * half + ((v >> 2) << 2);
      au0[v] = As[wm + mrow][ui];
      au1[v] = As[wm + 16 + mrow][ui];
      // B 32x16 layout: lane&15 -> N column, dword v -> K = 2*ui, 2*ui+1 (same K order as A)
      int nc = wn + mrow, sh = (nc & 1) << 4;
      bu0[v] = ((Bs[2 * ui][nc >> 1] >> sh) & 0xFFFFu) | (((Bs[2 * ui + 1][nc >> 1] >> sh) & 0xFFFFu) << 16);
      bu1[v] = ((Bs[2 * ui][(nc >> 1) + 8] >> sh) & 0xFFFFu) | (((Bs[2 * ui + 1][(nc >> 1) + 8] >> sh) & 0xFFFFu) << 16);
    }
    v16bf a0 = __builtin_bit_cast(v16bf, au0);
    v16bf a1 = __builtin_bit_cast(v16bf, au1);
    v16bf b0 = __builtin_bit_cast(v16bf, bu0);
    v16bf b1 = __builtin_bit_cast(v16bf, bu1);

    acc00 = wmma_bf16(a0, b0, acc00);
    acc01 = wmma_bf16(a0, b1, acc01);
    acc10 = wmma_bf16(a1, b0, acc10);
    acc11 = wmma_bf16(a1, b1, acc11);
    __syncthreads();
  }

  // C/D layout: dword v -> row v + 8*half, col = lane & 15
  float*  Cf = (float*)Cv + zo * sC0 + zi * sC1;
  u16b*   Cb = (u16b*) Cv + zo * sC0 + zi * sC1;
  auto stc = [&](v8f acc, int bm, int bn) {
    #pragma unroll
    for (int v = 0; v < 8; ++v) {
      int gm = bm + v + 8 * half;
      int gn = bn + mrow;
      if (gm < M && gn < N) {
        float val = acc[v];
        if (bias) val += bias[gn];
        if (relu) val = fmaxf(val, 0.0f);
        if (cbf) Cb[(long long)gm * ldc + gn] = f2bfu(val);
        else     Cf[(long long)gm * ldc + gn] = val;
      }
    }
  };
  stc(acc00, m0 + wm,      n0 + wn);
  stc(acc01, m0 + wm,      n0 + wn + 16);
  stc(acc10, m0 + wm + 16, n0 + wn);
  stc(acc11, m0 + wm + 16, n0 + wn + 16);
}

// ---------------- elementwise / reduction kernels ----------------
__global__ void cvt_kernel(u16b* o, const float* in, long long n) {
  long long i = (long long)blockIdx.x * blockDim.x + threadIdx.x;
  if (i < n) o[i] = f2bfu(in[i]);
}

__global__ void embed_kernel(float* h, const int* x, const float* emb) {
  long long idx = (long long)blockIdx.x * blockDim.x + threadIdx.x;
  if (idx >= (long long)ROWS_Q * D_) return;
  int c = (int)(idx % D_);
  long long r = idx / D_;                  // r = i*B + b, x is [QLEN,B] row-major
  int tok = x[r];
  h[idx] = emb[(long long)tok * D_ + c] * 27.712812921102035f;   // sqrt(768)
}

__global__ void posemb_kernel(u16b* r_) {
  long long idx = (long long)blockIdx.x * blockDim.x + threadIdx.x;
  if (idx >= (long long)KLEN_ * D_) return;
  int c = (int)(idx % D_);
  int p = (int)(idx / D_);
  float pos = (float)(KLEN_ - 1 - p);
  int j = (c < D_ / 2) ? c : (c - D_ / 2);
  float inv = __expf(-((2.0f * (float)j) / (float)D_) * 9.210340371976184f); // 10000^(-2j/D)
  float a = pos * inv;
  r_[idx] = f2bfu((c < D_ / 2) ? sinf(a) : cosf(a));
}

__global__ void cat_kernel(u16b* cat, const float* cond, const float* mem_l, const float* h) {
  long long idx = (long long)blockIdx.x * blockDim.x + threadIdx.x;
  if (idx >= (long long)KLEN_ * BSZ_ * D_) return;
  int c = (int)(idx % D_);
  long long r = idx / D_;
  int b = (int)(r % BSZ_);
  int kpos = (int)(r / BSZ_);
  float v;
  if (kpos < CLEN_)              v = cond[((long long)kpos * BSZ_ + b) * D_ + c];
  else if (kpos < CLEN_ + MLEN_) v = mem_l[((long long)(kpos - CLEN_) * BSZ_ + b) * D_ + c];
  else                           v = h[((long long)(kpos - CLEN_ - MLEN_) * BSZ_ + b) * D_ + c];
  cat[idx] = f2bfu(v);
}

__global__ void qbias_kernel(u16b* qw, u16b* qr, const float* heads,
                             const float* rwb, const float* rrb) {
  long long idx = (long long)blockIdx.x * blockDim.x + threadIdx.x;
  if (idx >= (long long)BSZ_ * NH_ * QLEN_ * DH_) return;
  int d = (int)(idx % DH_);
  long long r = idx / DH_;
  int i = (int)(r % QLEN_); r /= QLEN_;
  int n = (int)(r % NH_);
  int b = (int)(r / NH_);
  float src = heads[(long long)((CLEN_ + MLEN_ + i) * BSZ_ + b) * QKV3_ + n * DH_ + d];
  qw[idx] = f2bfu(src + rwb[n * DH_ + d]);
  qr[idx] = f2bfu(src + rrb[n * DH_ + d]);
}

// khT[b][n][d][j] = kh[j][b][n][d]   (makes AC a plain NN GEMM)
__global__ void transpose_k_kernel(u16b* khT, const float* heads) {
  long long idx = (long long)blockIdx.x * blockDim.x + threadIdx.x;
  if (idx >= (long long)BSZ_ * NH_ * DH_ * KLEN_) return;
  int j = (int)(idx % KLEN_);
  long long r = idx / KLEN_;
  int d = (int)(r % DH_); r /= DH_;
  int n = (int)(r % NH_);
  int b = (int)(r / NH_);
  khT[idx] = f2bfu(heads[((long long)j * BSZ_ + b) * QKV3_ + HDIM_ + n * DH_ + d]);
}

// rkT[n][d][j] = rk[j][n][d]
__global__ void transpose_rk_kernel(u16b* rkT, const float* rk) {
  long long idx = (long long)blockIdx.x * blockDim.x + threadIdx.x;
  if (idx >= (long long)HDIM_ * KLEN_) return;
  int j  = (int)(idx % KLEN_);
  int nd = (int)(idx / KLEN_);
  rkT[idx] = f2bfu(rk[(long long)j * HDIM_ + nd]);
}

// vh_bf[j*B+b][n*64+d] = bf16(heads[j*B+b][1536 + n*64+d])
__global__ void cvt_v_kernel(u16b* vh, const float* heads) {
  long long idx = (long long)blockIdx.x * blockDim.x + threadIdx.x;
  if (idx >= (long long)ROWS_K * HDIM_) return;
  int nd = (int)(idx % HDIM_);
  long long row = idx / HDIM_;
  vh[idx] = f2bfu(heads[row * QKV3_ + 2 * HDIM_ + nd]);
}

// score = mask ? -inf : (AC + rel_shift(BD)) / sqrt(DH); written in-place into sc
__global__ void combine_kernel(float* sc, const float* __restrict__ bd) {
  long long idx = (long long)blockIdx.x * blockDim.x + threadIdx.x;
  if (idx >= (long long)BSZ_ * NH_ * QLEN_ * KLEN_) return;
  int j = (int)(idx % KLEN_);
  long long r = idx / KLEN_;
  int i = (int)(r % QLEN_);
  long long z = r / QLEN_;
  float v;
  if (j > i + CLEN_ + MLEN_) {
    v = -__builtin_inff();
  } else {
    int jj = j + (QLEN_ - 1 - i);                       // rel_shift gather
    float bdv = (jj < KLEN_) ? bd[(z * QLEN_ + i) * (long long)KLEN_ + jj] : 0.0f;
    v = (sc[idx] + bdv) * 0.125f;                       // 1/sqrt(64)
  }
  sc[idx] = v;
}

__global__ __launch_bounds__(128) void softmax_kernel(float* sc, u16b* prob) {
  __shared__ float red[128];
  float* p  = sc   + (long long)blockIdx.x * KLEN_;
  u16b*  pb = prob + (long long)blockIdx.x * KLEN_;
  const int t = threadIdx.x;
  float m = -__builtin_inff();
  for (int j = t; j < KLEN_; j += 128) m = fmaxf(m, p[j]);
  red[t] = m; __syncthreads();
  for (int off = 64; off > 0; off >>= 1) { if (t < off) red[t] = fmaxf(red[t], red[t + off]); __syncthreads(); }
  m = red[0]; __syncthreads();
  float s = 0.0f;
  for (int j = t; j < KLEN_; j += 128) { float e = __expf(p[j] - m); p[j] = e; s += e; }
  red[t] = s; __syncthreads();
  for (int off = 64; off > 0; off >>= 1) { if (t < off) red[t] += red[t + off]; __syncthreads(); }
  float inv = 1.0f / red[0];
  for (int j = t; j < KLEN_; j += 128) pb[j] = f2bfu(p[j] * inv);
}

// h = LayerNorm(x + y) * g + b ; also emits bf16 copy (row = blockIdx.x; in-place safe)
__global__ __launch_bounds__(256) void addln_kernel(const float* __restrict__ x,
                                                    const float* __restrict__ y,
                                                    const float* __restrict__ g,
                                                    const float* __restrict__ bta,
                                                    float* out, u16b* out_bf) {
  __shared__ float red[256];
  const long long row = blockIdx.x;
  const int t = threadIdx.x;
  float v[3]; float s = 0.0f;
  #pragma unroll
  for (int e = 0; e < 3; ++e) {
    int c = t + e * 256;
    v[e] = x[row * D_ + c] + y[row * D_ + c];
    s += v[e];
  }
  red[t] = s; __syncthreads();
  for (int off = 128; off > 0; off >>= 1) { if (t < off) red[t] += red[t + off]; __syncthreads(); }
  float mean = red[0] * (1.0f / D_);
  __syncthreads();
  s = 0.0f;
  #pragma unroll
  for (int e = 0; e < 3; ++e) { float d = v[e] - mean; s += d * d; }
  red[t] = s; __syncthreads();
  for (int off = 128; off > 0; off >>= 1) { if (t < off) red[t] += red[t + off]; __syncthreads(); }
  float rstd = rsqrtf(red[0] * (1.0f / D_) + 1e-5f);
  #pragma unroll
  for (int e = 0; e < 3; ++e) {
    int c = t + e * 256;
    float o = (v[e] - mean) * rstd * g[c] + bta[c];
    out[row * D_ + c] = o;
    out_bf[row * D_ + c] = f2bfu(o);
  }
}

// ---------------- host orchestration ----------------
static inline dim3 gblk(long long n) { return dim3((unsigned)((n + 255) / 256)); }

extern "C" void kernel_launch(void* const* d_in, const int* in_sizes, int n_in,
                              void* d_out, int out_size, void* d_ws, size_t ws_size,
                              hipStream_t stream) {
  (void)in_sizes; (void)n_in; (void)out_size; (void)ws_size;
  const int*   x       = (const int*)  d_in[0];
  const float* cond    = (const float*)d_in[1];
  const float* mems    = (const float*)d_in[2];
  const float* emb     = (const float*)d_in[3];
  const float* qkv_w   = (const float*)d_in[4];
  const float* r_net_w = (const float*)d_in[5];
  const float* o_w     = (const float*)d_in[6];
  const float* ln1_g   = (const float*)d_in[7];
  const float* ln1_b   = (const float*)d_in[8];
  const float* w1      = (const float*)d_in[9];
  const float* b1      = (const float*)d_in[10];
  const float* w2      = (const float*)d_in[11];
  const float* b2      = (const float*)d_in[12];
  const float* ln2_g   = (const float*)d_in[13];
  const float* ln2_b   = (const float*)d_in[14];
  const float* rwb     = (const float*)d_in[15];
  const float* rrb     = (const float*)d_in[16];
  const float* proj_w  = (const float*)d_in[17];
  const float* proj_b  = (const float*)d_in[18];
  float* out = (float*)d_out;

  // ---- workspace layout: f32 region first, then bf16 (u16) region (16B aligned)
  float* wsf = (float*)d_ws;
  float* heads = wsf;  wsf += (long long)ROWS_K * QKV3_;
  float* rk    = wsf;  wsf += (long long)KLEN_ * HDIM_;
  float* sc    = wsf;  wsf += (long long)BSZ_ * NH_ * QLEN_ * KLEN_;
  float* bd    = wsf;  wsf += (long long)BSZ_ * NH_ * QLEN_ * KLEN_;
  float* ob    = wsf;  wsf += (long long)ROWS_Q * D_;
  float* h     = wsf;  wsf += (long long)ROWS_Q * D_;
  float* ff2   = wsf;  wsf += (long long)ROWS_Q * D_;

  u16b* wsh = (u16b*)wsf;
  u16b* cat_bf  = wsh;  wsh += (long long)KLEN_ * BSZ_ * D_;
  u16b* r_bf    = wsh;  wsh += (long long)KLEN_ * D_;
  u16b* qw_bf   = wsh;  wsh += (long long)BSZ_ * NH_ * QLEN_ * DH_;
  u16b* qr_bf   = wsh;  wsh += (long long)BSZ_ * NH_ * QLEN_ * DH_;
  u16b* khT_bf  = wsh;  wsh += (long long)BSZ_ * NH_ * DH_ * KLEN_;
  u16b* rkT_bf  = wsh;  wsh += (long long)HDIM_ * KLEN_;
  u16b* prob_bf = wsh;  wsh += (long long)BSZ_ * NH_ * QLEN_ * KLEN_;
  u16b* vh_bf   = wsh;  wsh += (long long)ROWS_K * HDIM_;
  u16b* av_bf   = wsh;  wsh += (long long)ROWS_Q * D_;
  u16b* ff1_bf  = wsh;  wsh += (long long)ROWS_Q * DI_;
  u16b* h_bf    = wsh;  wsh += (long long)ROWS_Q * D_;
  u16b* wq_bf   = wsh;  wsh += (long long)D_ * QKV3_;     // per-layer reused weight buffers
  u16b* wr_bf   = wsh;  wsh += (long long)D_ * HDIM_;
  u16b* wo_bf   = wsh;  wsh += (long long)HDIM_ * D_;
  u16b* w1_bf   = wsh;  wsh += (long long)D_ * DI_;
  u16b* w2_bf   = wsh;  wsh += (long long)DI_ * D_;
  u16b* wp_bf   = wsh;  wsh += (long long)D_ * V_;

  auto cvt = [&](u16b* dst, const float* src, long long n) {
    cvt_kernel<<<gblk(n), 256, 0, stream>>>(dst, src, n);
  };
  auto gemm = [&](const u16b* A, long long sA0, long long sA1, int lda,
                  const u16b* B, long long sB0, long long sB1, int ldb,
                  void* C, long long sC0, long long sC1, int ldc,
                  int M, int N, int K, int nz, int zdiv,
                  const float* bias, int relu, int cbf) {
    dim3 g((unsigned)((N + 63) / 64), (unsigned)((M + 127) / 128), (unsigned)nz);
    gemm_wmma<<<g, 256, 0, stream>>>(A, sA0, sA1, lda, B, sB0, sB1, ldb,
                                     C, sC0, sC1, ldc, M, N, K, zdiv, bias, relu, cbf);
  };

  embed_kernel<<<gblk((long long)ROWS_Q * D_), 256, 0, stream>>>(h, x, emb);
  posemb_kernel<<<gblk((long long)KLEN_ * D_), 256, 0, stream>>>(r_bf);

  const long long sQA0 = (long long)NH_ * QLEN_ * DH_,   sQA1 = (long long)QLEN_ * DH_;
  const long long sSC0 = (long long)NH_ * QLEN_ * KLEN_, sSC1 = (long long)QLEN_ * KLEN_;
  const long long sKT0 = (long long)NH_ * DH_ * KLEN_,   sKT1 = (long long)DH_ * KLEN_;

  for (int l = 0; l < L_; ++l) {
    // per-layer weight conversion into reused bf16 buffers
    cvt(wq_bf, qkv_w   + (long long)l * D_ * QKV3_, (long long)D_ * QKV3_);
    cvt(wr_bf, r_net_w + (long long)l * D_ * HDIM_, (long long)D_ * HDIM_);
    cvt(wo_bf, o_w     + (long long)l * HDIM_ * D_, (long long)HDIM_ * D_);
    cvt(w1_bf, w1      + (long long)l * D_ * DI_,   (long long)D_ * DI_);
    cvt(w2_bf, w2      + (long long)l * DI_ * D_,   (long long)DI_ * D_);

    const float* mem_l = mems + (long long)l * MLEN_ * BSZ_ * D_;
    cat_kernel<<<gblk((long long)KLEN_ * BSZ_ * D_), 256, 0, stream>>>(cat_bf, cond, mem_l, h);

    // heads = cat @ qkv_w[l]           [4224 x 2304]
    gemm(cat_bf, 0, 0, D_, wq_bf, 0, 0, QKV3_,
         heads, 0, 0, QKV3_, ROWS_K, QKV3_, D_, 1, 1, nullptr, 0, 0);
    // rk = r @ r_net_w[l]              [1056 x 768]
    gemm(r_bf, 0, 0, D_, wr_bf, 0, 0, HDIM_,
         rk, 0, 0, HDIM_, KLEN_, HDIM_, D_, 1, 1, nullptr, 0, 0);

    qbias_kernel<<<gblk((long long)BSZ_ * NH_ * QLEN_ * DH_), 256, 0, stream>>>(qw_bf, qr_bf, heads, rwb, rrb);
    transpose_k_kernel<<<gblk((long long)BSZ_ * NH_ * DH_ * KLEN_), 256, 0, stream>>>(khT_bf, heads);
    transpose_rk_kernel<<<gblk((long long)HDIM_ * KLEN_), 256, 0, stream>>>(rkT_bf, rk);
    cvt_v_kernel<<<gblk((long long)ROWS_K * HDIM_), 256, 0, stream>>>(vh_bf, heads);

    // AC[b,n] = (q + r_w_bias) @ khT   per head: [512 x 1056], K = 64
    gemm(qw_bf, sQA0, sQA1, DH_,
         khT_bf, sKT0, sKT1, KLEN_,
         sc, sSC0, sSC1, KLEN_, QLEN_, KLEN_, DH_, BSZ_ * NH_, NH_, nullptr, 0, 0);
    // rawBD[b,n] = (q + r_r_bias) @ rkT
    gemm(qr_bf, sQA0, sQA1, DH_,
         rkT_bf, 0, sKT1, KLEN_,
         bd, sSC0, sSC1, KLEN_, QLEN_, KLEN_, DH_, BSZ_ * NH_, NH_, nullptr, 0, 0);

    combine_kernel<<<gblk((long long)BSZ_ * NH_ * QLEN_ * KLEN_), 256, 0, stream>>>(sc, bd);
    softmax_kernel<<<dim3(BSZ_ * NH_ * QLEN_), 128, 0, stream>>>(sc, prob_bf);

    // av[i,b,n,d] = prob @ v  (bf16 C)  per head: [512 x 64], K = 1056
    gemm(prob_bf, sSC0, sSC1, KLEN_,
         vh_bf, (long long)HDIM_, (long long)DH_, BSZ_ * HDIM_,
         av_bf, (long long)D_, (long long)DH_, BSZ_ * D_,
         QLEN_, DH_, KLEN_, BSZ_ * NH_, NH_, nullptr, 0, 1);

    // attn out projection + LN1
    gemm(av_bf, 0, 0, D_, wo_bf, 0, 0, D_,
         ob, 0, 0, D_, ROWS_Q, D_, HDIM_, 1, 1, nullptr, 0, 0);
    addln_kernel<<<dim3(ROWS_Q), 256, 0, stream>>>(h, ob, ln1_g + l * D_, ln1_b + l * D_, h, h_bf);

    // FFN + LN2
    gemm(h_bf, 0, 0, D_, w1_bf, 0, 0, DI_,
         ff1_bf, 0, 0, DI_, ROWS_Q, DI_, D_, 1, 1, b1 + (long long)l * DI_, 1, 1);
    gemm(ff1_bf, 0, 0, DI_, w2_bf, 0, 0, D_,
         ff2, 0, 0, D_, ROWS_Q, D_, DI_, 1, 1, b2 + l * D_, 0, 0);
    addln_kernel<<<dim3(ROWS_Q), 256, 0, stream>>>(h, ff2, ln2_g + l * D_, ln2_b + l * D_, h, h_bf);
  }

  // logits = h @ proj_w + proj_b       [2048 x 10000]
  cvt(wp_bf, proj_w, (long long)D_ * V_);
  gemm(h_bf, 0, 0, D_, wp_bf, 0, 0, V_, out, 0, 0, V_,
       ROWS_Q, V_, D_, 1, 1, proj_b, 0, 0);
}
